// QPLEXQattenWeight_89404039234253
// MI455X (gfx1250) — compile-verified, matrix operand order checked by reading, not run on hardware
//
#include <hip/hip_runtime.h>

// ---------------- problem constants ----------------
#define Bv    16384   // batch
#define Av    64      // agents
#define Uv    64      // unit dim
#define Sv    4096    // state dim
#define Ev    32      // embed
#define HYPv  64      // hypernet hidden
#define Hv    4       // heads
#define NP    384     // 352 fused-GEMM columns padded to 384 (24 WMMA tiles)
#define KT    32      // K step per WMMA
#define INV_SQRT_E 0.17677669529663689f
#define REGC  1e-3f
#define ATT_BLOCKS 2048

typedef __attribute__((ext_vector_type(16))) __bf16 bf16x16;
typedef __attribute__((ext_vector_type(8)))  float  f32x8;

union FragAB { uint4 u[2]; __bf16 h[16]; bf16x16 v; };
union FragCD { f32x8 v; float f[8]; };

// ---------------- pack: Wcat^T (bf16, [NP][Sv]) ----------------
// columns: [0,256) sel_W1[h][k][c] (h=n>>6,c=n&63); [256,320) wh_W1; [320,352) v_W1; [352,384) zero
__global__ __launch_bounds__(256)
void pack_w_kernel(const float* __restrict__ sel_W1, const float* __restrict__ wh_W1,
                   const float* __restrict__ v_W1, __bf16* __restrict__ WcatT) {
  const int idx = blockIdx.x * 256 + threadIdx.x;
  if (idx >= NP * Sv) return;
  const int n = idx >> 12;          // 0..383
  const int k = idx & (Sv - 1);     // 0..4095
  float v = 0.0f;
  if (n < 256) {
    const int h = n >> 6, c = n & 63;
    v = sel_W1[((size_t)h * Sv + k) * HYPv + c];
  } else if (n < 320) {
    v = wh_W1[(size_t)k * HYPv + (n - 256)];
  } else if (n < 352) {
    v = v_W1[(size_t)k * Ev + (n - 320)];
  }
  WcatT[(size_t)n * Sv + k] = (__bf16)v;
}

// ---------------- pack: M[h][u][c] = sum_e key_W[h,u,e]*sel_W2[h,c,e] ----------------
__global__ __launch_bounds__(256)
void pack_m_kernel(const float* __restrict__ key_W, const float* __restrict__ sel_W2,
                   float* __restrict__ Mmat) {
  const int idx = blockIdx.x * 256 + threadIdx.x;   // h*4096 + u*64 + c
  if (idx >= Hv * Uv * HYPv) return;
  const int h = idx >> 12, u = (idx >> 6) & 63, c = idx & 63;
  const float* kw = key_W  + ((size_t)h * Uv  + u) * Ev;
  const float* sw = sel_W2 + ((size_t)h * HYPv + c) * Ev;
  float acc = 0.0f;
  for (int e = 0; e < Ev; ++e) acc += kw[e] * sw[e];
  Mmat[idx] = acc;
}

// ---------------- fused GEMM: out1[B][NP] = states[B][Sv] x Wcat (bf16 WMMA) ----------------
// 8 waves/WG; WG owns 32 rows. wave: mhalf = w&1 (16-row half), nset = w>>1 (6 n-tiles).
__global__ __launch_bounds__(256)
void gemm1_kernel(const float* __restrict__ states, const __bf16* __restrict__ WcatT,
                  float* __restrict__ out1) {
  __shared__ __align__(16) __bf16 As[32][40];    // 32 rows x 32 k (+pad)
  __shared__ __align__(16) __bf16 Bs[NP][40];    // 384 cols x 32 k (+pad), transposed
  const int tid  = threadIdx.x;
  const int wave = tid >> 5, lane = tid & 31;
  const int mhalf = wave & 1;
  const int nset  = wave >> 1;
  const int g  = lane >> 4;        // lane group per ISA layout
  const int mn = lane & 15;        // row (A/D) or column (B/D) index
  const int b0 = blockIdx.x * 32;

  const f32x8 zero = {0.f, 0.f, 0.f, 0.f, 0.f, 0.f, 0.f, 0.f};
  f32x8 acc[6];
  for (int j = 0; j < 6; ++j) acc[j] = zero;

  for (int k0 = 0; k0 < Sv; k0 += KT) {
    { // A tile: 32x32 fp32 -> bf16 (one float4 per thread)
      const int r  = tid >> 3;
      const int c4 = (tid & 7) * 4;
      const float4 f = *(const float4*)&states[(size_t)(b0 + r) * Sv + k0 + c4];
      As[r][c4 + 0] = (__bf16)f.x;  As[r][c4 + 1] = (__bf16)f.y;
      As[r][c4 + 2] = (__bf16)f.z;  As[r][c4 + 3] = (__bf16)f.w;
    }
    // B tile: 384 cols x 32 k bf16, 16-byte chunks (6 per thread)
    for (int j = 0; j < 6; ++j) {
      const int c  = tid + j * 256;        // chunk id, 1536 total
      const int n  = c >> 2;
      const int kc = (c & 3) * 8;
      *(uint4*)&Bs[n][kc] = *(const uint4*)&WcatT[(size_t)n * Sv + k0 + kc];
    }
    __syncthreads();

    // A fragment: a[i]=A[m][g*8+i], a[8+i]=A[m][16+g*8+i]
    FragAB af;
    {
      const int row = mhalf * 16 + mn;
      af.u[0] = *(const uint4*)&As[row][g * 8];
      af.u[1] = *(const uint4*)&As[row][16 + g * 8];
    }
    for (int j = 0; j < 6; ++j) {
      const int n = nset * 96 + j * 16 + mn;
      FragAB bf;
      bf.u[0] = *(const uint4*)&Bs[n][g * 8];
      bf.u[1] = *(const uint4*)&Bs[n][16 + g * 8];
      acc[j] = __builtin_amdgcn_wmma_f32_16x16x32_bf16(
          false, af.v, false, bf.v, (short)0, acc[j], false, false);
    }
    __syncthreads();
  }

  // D layout: lane (g, n=mn): d[r] = C[g*8+r][n]
  for (int j = 0; j < 6; ++j) {
    FragCD cd; cd.v = acc[j];
    const int col  = nset * 96 + j * 16 + mn;
    const int rowb = b0 + mhalf * 16 + g * 8;
    for (int r = 0; r < 8; ++r)
      out1[(size_t)(rowb + r) * NP + col] = cd.f[r];
  }
}

// ---------------- per-batch head math: q, |w_head|, V(s) ----------------
__global__ __launch_bounds__(256)
void head_kernel(const float* __restrict__ out1, const float* __restrict__ Mmat,
                 const float* __restrict__ sel_b1, const float* __restrict__ wh_b1,
                 const float* __restrict__ v_b1, const float* __restrict__ wh_W2,
                 const float* __restrict__ wh_b2, const float* __restrict__ v_W2,
                 const float* __restrict__ v_b2, float* __restrict__ q,
                 float* __restrict__ whd, float* __restrict__ vout) {
  __shared__ float r[8][256];
  const int tid = threadIdx.x, wave = tid >> 5, lane = tid & 31;
  const int b = blockIdx.x * 8 + wave;
  const float* row = out1 + (size_t)b * NP;
  for (int i = 0; i < 8; ++i) {
    const int j = lane * 8 + i;
    const float x = row[j] + sel_b1[j];
    r[wave][j] = x > 0.f ? x : 0.f;
  }
  __syncthreads();
  // q[b,h,u] = sum_c M[h][u][c] * relu(sel_h[b,h,c]); lane does u=lane, lane+32
  for (int h = 0; h < Hv; ++h)
    for (int uu = 0; uu < 2; ++uu) {
      const int u = lane + uu * 32;
      const float* Mr = Mmat + ((size_t)h * 64 + u) * 64;
      float acc = 0.f;
      for (int c = 0; c < 64; ++c) acc += Mr[c] * r[wave][h * 64 + c];
      q[(size_t)b * 256 + h * 64 + u] = acc;
    }
  if (lane < Hv) {  // w_head[b,h] = |relu(wh hidden) @ wh_W2 + wh_b2|
    float acc = wh_b2[lane];
    for (int k = 0; k < HYPv; ++k) {
      float x = row[256 + k] + wh_b1[k];
      x = x > 0.f ? x : 0.f;
      acc += x * wh_W2[k * Hv + lane];
    }
    whd[(size_t)b * Hv + lane] = fabsf(acc);
  }
  if (lane == 8) {  // v[b]
    float acc = v_b2[0];
    for (int e = 0; e < Ev; ++e) {
      float x = row[320 + e] + v_b1[e];
      x = x > 0.f ? x : 0.f;
      acc += x * v_W2[e];
    }
    vout[b] = acc;
  }
}

// ---------------- logits, softmax, head_attend, reg + entropy partials ----------------
__global__ __launch_bounds__(256)
void attend_kernel(const float* __restrict__ states, const float* __restrict__ q,
                   const float* __restrict__ whd, float* __restrict__ ha,
                   float* __restrict__ part) {
  __shared__ __align__(16) float su[Sv];
  __shared__ float sq[256], slog[256], sw[256], red[256];
  __shared__ float shd[4], smax[4], ssum[4], ent_acc[4];
  __shared__ float sq_acc;
  const int tid = threadIdx.x;
  if (tid < 4) ent_acc[tid] = 0.f;
  if (tid == 0) sq_acc = 0.f;
  const int h = tid >> 6, a = tid & 63;

  for (int ib = 0; ib < 8; ++ib) {
    const int b = blockIdx.x * 8 + ib;
    for (int j = 0; j < 4; ++j) {
      const int o = (tid + j * 256) * 4;
      *(float4*)&su[o] = *(const float4*)&states[(size_t)b * Sv + o];
    }
    sq[tid] = q[(size_t)b * 256 + tid];
    if (tid < 4) shd[tid] = whd[(size_t)b * Hv + tid];
    __syncthreads();

    // logits[h][a] = sum_u su[a*64+u] * q[h][u]  (rotated u -> conflict-free banks)
    float lg = 0.f;
    for (int uu = 0; uu < 64; ++uu) {
      const int u = (a + uu) & 63;
      lg += su[a * 64 + u] * sq[h * 64 + u];
    }
    slog[tid] = lg;
    red[tid] = lg * lg;
    __syncthreads();

    if (tid == 0) { float s = 0.f; for (int i = 0; i < 256; ++i) s += red[i]; sq_acc += s; }
    if (tid < 4) {
      float m = -3.4e38f;
      for (int i = 0; i < 64; ++i) m = fmaxf(m, slog[tid * 64 + i]);
      smax[tid] = m;
    }
    __syncthreads();

    sw[tid] = __expf((slog[tid] - smax[h]) * INV_SQRT_E);
    __syncthreads();

    if (tid < 4) { float s = 0.f; for (int i = 0; i < 64; ++i) s += sw[tid * 64 + i]; ssum[tid] = s; }
    __syncthreads();

    const float w = sw[tid] / ssum[h];
    sw[tid] = w;
    red[tid] = w * __logf(w + 1e-8f);
    __syncthreads();

    if (tid < 4) { float s = 0.f; for (int i = 0; i < 64; ++i) s += red[tid * 64 + i]; ent_acc[tid] += s; }
    if (tid < 64) {  // head_attend[b,a] = sum_h w[h,a]*w_head[h]
      float s = 0.f;
      for (int h2 = 0; h2 < Hv; ++h2) s += sw[h2 * 64 + tid] * shd[h2];
      ha[(size_t)b * Av + tid] = s;
    }
    __syncthreads();
  }
  if (tid == 0) part[(size_t)blockIdx.x * 5] = sq_acc;
  if (tid < 4)  part[(size_t)blockIdx.x * 5 + 1 + tid] = ent_acc[tid];
}

// ---------------- final scalar reduction ----------------
__global__ void finalize_kernel(const float* __restrict__ part,
                                float* __restrict__ outreg, float* __restrict__ outent) {
  const int t = threadIdx.x;
  if (t < 5) {
    float s = 0.f;
    for (int i = 0; i < ATT_BLOCKS; ++i) s += part[(size_t)i * 5 + t];
    if (t == 0) outreg[0] = REGC * s / ((float)Bv * (float)Av);
    else        outent[t - 1] = -s / (float)Bv;
  }
}

// ---------------- launch ----------------
extern "C" void kernel_launch(void* const* d_in, const int* in_sizes, int n_in,
                              void* d_out, int out_size, void* d_ws, size_t ws_size,
                              hipStream_t stream) {
  const float* states = (const float*)d_in[1];
  const float* sel_W1 = (const float*)d_in[2];
  const float* sel_b1 = (const float*)d_in[3];
  const float* sel_W2 = (const float*)d_in[4];
  const float* key_W  = (const float*)d_in[5];
  const float* wh_W1  = (const float*)d_in[6];
  const float* wh_b1  = (const float*)d_in[7];
  const float* wh_W2  = (const float*)d_in[8];
  const float* wh_b2  = (const float*)d_in[9];
  const float* v_W1   = (const float*)d_in[10];
  const float* v_b1   = (const float*)d_in[11];
  const float* v_W2   = (const float*)d_in[12];
  const float* v_b2   = (const float*)d_in[13];

  char* ws = (char*)d_ws;
  __bf16* WcatT = (__bf16*)(ws + 0);              // 384*4096*2  = 3,145,728
  float*  Mmat  = (float*)(ws + 3145728);         // 4*64*64*4   = 65,536
  float*  out1  = (float*)(ws + 3211264);         // 16384*384*4 = 25,165,824
  float*  qbuf  = (float*)(ws + 28377088);        // 16384*256*4 = 16,777,216
  float*  whd   = (float*)(ws + 45154304);        // 16384*4*4   = 262,144
  float*  part  = (float*)(ws + 45416448);        // 2048*5*4    = 40,960

  float* out    = (float*)d_out;
  float* ha     = out;                            // [B*A]
  float* vout   = out + (size_t)Bv * Av;          // [B]
  float* outreg = vout + Bv;                      // [1]
  float* outent = outreg + 1;                     // [H]

  pack_w_kernel<<<(NP * Sv + 255) / 256, 256, 0, stream>>>(sel_W1, wh_W1, v_W1, WcatT);
  pack_m_kernel<<<(Hv * Uv * HYPv + 255) / 256, 256, 0, stream>>>(key_W, sel_W2, Mmat);
  gemm1_kernel<<<Bv / 32, 256, 0, stream>>>(states, WcatT, out1);
  head_kernel<<<Bv / 8, 256, 0, stream>>>(out1, Mmat, sel_b1, wh_b1, v_b1,
                                          wh_W2, wh_b2, v_W2, v_b2, qbuf, whd, vout);
  attend_kernel<<<ATT_BLOCKS, 256, 0, stream>>>(states, qbuf, whd, ha, part);
  finalize_kernel<<<1, 32, 0, stream>>>(part, outreg, outent);
}
